// TransformerDecoderLayer_63737314673206
// MI455X (gfx1250) — compile-verified, matrix operand order checked by reading, not run on hardware
//
#include <hip/hip_runtime.h>
#include <hip/hip_bf16.h>
#include <cstdint>
#include <cstddef>

// ---------------------------------------------------------------------------
// Shapes (fixed by the reference setup): C=256, heads=8, dh=32, LQ=64,
// NSEL=256, B=4, H=W=128, win=16x16, q grid 64x64 with 8x8 query windows,
// D_FFN=1024. win_h/win_w/q_h/q_w scalar inputs are therefore hardcoded.
// ---------------------------------------------------------------------------

typedef __attribute__((ext_vector_type(16))) __bf16        v16bf;
typedef __attribute__((ext_vector_type(8)))  float         v8f;
typedef __attribute__((ext_vector_type(4)))  int           v4i;
typedef __attribute__((ext_vector_type(8)))  int           v8i;
typedef __attribute__((ext_vector_type(4)))  unsigned int  v4u;

union V16BF { v16bf v; unsigned short u[16]; };

__device__ __forceinline__ unsigned short f2bf(float x) {
  unsigned int b = __float_as_uint(x);
  unsigned int r = b + 0x7FFFu + ((b >> 16) & 1u);   // round-to-nearest-even
  return (unsigned short)(r >> 16);
}

// CDNA5 ISA §7.12.2: 16-bit A-matrix 16x32 per-lane K indices.
// lanes 0-15: v0:K0,1 v1:K2,3 v2:K4,5 v3:K6,7 v4:K16,17 ... v7:K22,23
// lanes 16-31: +8.  B (32x16) uses the symmetric lane->N, element->K map.
__device__ __forceinline__ int wmma_kidx(int lane, int e) {
  int v = e >> 1;
  int base = (v < 4) ? (v << 1) : (16 + ((v - 4) << 1));
  return base + (e & 1) + ((lane & 16) ? 8 : 0);
}

__device__ __forceinline__ v8f wmma_bf16(v16bf a, v16bf b, v8f c) {
  return __builtin_amdgcn_wmma_f32_16x16x32_bf16(false, a, false, b,
                                                 (short)0, c, false, false);
}

// ---------------------------------------------------------------------------
// CDNA5 data movers, feature-guarded:
//  - TDM TENSOR_LOAD_TO_LDS (ISA 08, D# groups per §8.3/8.4), TENSORcnt
//  - GLOBAL_LOAD_ASYNC_TO_LDS_B128, ASYNCcnt
//  - plain float4 LDS stores as last resort
// ---------------------------------------------------------------------------
#if defined(__gfx1250__) && \
    __has_builtin(__builtin_amdgcn_global_load_async_to_lds_b128) && \
    __has_builtin(__builtin_amdgcn_s_wait_asynccnt)
#define USE_ASYNC_LDS 1
#else
#define USE_ASYNC_LDS 0
#endif

#if defined(__gfx1250__) && \
    __has_builtin(__builtin_amdgcn_tensor_load_to_lds) && \
    __has_builtin(__builtin_amdgcn_s_wait_tensorcnt)
#define USE_TDM 1
#else
#define USE_TDM 0
#endif

__device__ __forceinline__ void copy16_g2l(const void* gsrc, void* ldst) {
#if USE_ASYNC_LDS
  __attribute__((address_space(1))) v4i* gp =
      (__attribute__((address_space(1))) v4i*)(unsigned long long)gsrc;
  __attribute__((address_space(3))) v4i* lp =
      (__attribute__((address_space(3))) v4i*)(unsigned int)(unsigned long long)ldst;
  __builtin_amdgcn_global_load_async_to_lds_b128(gp, lp, 0, 0);
#else
  *(float4*)ldst = *(const float4*)gsrc;
#endif
}

__device__ __forceinline__ void wait_async_lds() {
#if USE_ASYNC_LDS
  __builtin_amdgcn_s_wait_asynccnt(0);
#endif
}

#if USE_TDM
// DMA a 2D tile of fp32 (rows x 32 elems, row stride = stride_elems) into LDS
// laid out with a 4-DWORD pad after every 32-DWORD row (-> float[rows][36]).
// D# group0: count=1 | lds_addr | global_addr[56:0] | type=2.
// D# group1: data_size=4B, pad_enable, pad_interval=32 dwords (code 4),
//            pad_amount=4 dwords (code 3), tensor=tile dims, dim0 stride.
__device__ __forceinline__ void tdm_load_tile32(const float* gsrc, int stride_elems,
                                                void* lds_dst, int rows) {
  unsigned long long ga = (unsigned long long)gsrc;
  unsigned int la = (unsigned int)(unsigned long long)lds_dst;
  v4u g0;
  g0[0] = 0x1u;                                                  // count=1
  g0[1] = la;                                                    // lds_addr
  g0[2] = (unsigned int)ga;                                      // gaddr[31:0]
  g0[3] = ((unsigned int)(ga >> 32) & 0x01FFFFFFu) | 0x80000000u; // gaddr[56:32]|type=2
  const unsigned int td0 = 32u;                // tensor_dim0 (elements per row)
  const unsigned int td1 = (unsigned int)rows; // tensor_dim1 (rows)
  const unsigned int st0 = (unsigned int)stride_elems;
  v8i g1;
  g1[0] = (int)((2u << 16) | (1u << 20) | (4u << 22) | (3u << 25));
  g1[1] = (int)((td0 & 0xFFFFu) << 16);                 // tensor_dim0 lo
  g1[2] = (int)((td0 >> 16) | ((td1 & 0xFFFFu) << 16)); // dim0 hi | dim1 lo
  g1[3] = (int)((td1 >> 16) | (32u << 16));             // dim1 hi | tile_dim0=32
  g1[4] = rows;                                         // tile_dim1 (tile_dim2=0)
  g1[5] = (int)st0;                                     // dim0_stride lo
  g1[6] = 0;                                            // dim0_stride hi | dim1_stride lo
  g1[7] = 0;
  v4i g2 = {0, 0, 0, 0};
  v4i g3 = {0, 0, 0, 0};
#if defined(__clang_major__) && (__clang_major__ >= 23)
  v8i g4 = {0, 0, 0, 0, 0, 0, 0, 0};
  __builtin_amdgcn_tensor_load_to_lds(g0, g1, g2, g3, g4, 0);
#else
  __builtin_amdgcn_tensor_load_to_lds(g0, g1, g2, g3, 0);
#endif
}
#endif  // USE_TDM

// ---------------------------------------------------------------------------
// Generic GEMM: C[M,N] = act(A[M,K] @ W[N,K]^T + bias), fp32 in/out,
// bf16 WMMA compute with fp32 accumulation.
// M%128==0, N%64==0, K%32==0 (true for every call here).
// Block = 256 threads (8 waves); block tile 128x64; each wave owns a 32x32
// C tile -> 2 A-frags x 2 B-frags -> 4 WMMAs per 32-K step.
// Tiles staged to LDS as raw fp32 via TDM (or async-DMA fallback);
// fp32->bf16 conversion happens at fragment build.
// ---------------------------------------------------------------------------
__global__ void __launch_bounds__(256)
gemm_bias_act(const float* __restrict__ A, const float* __restrict__ W,
              const float* __restrict__ bias, float* __restrict__ C,
              int M, int N, int K, int act) {
  __shared__ __align__(16) float Asf[128][36];  // 32 used + 4 pad DWORDs
  __shared__ __align__(16) float Bsf[64][36];
  const int bm = blockIdx.y << 7;
  const int bn = blockIdx.x << 6;
  const int tid = threadIdx.x;
  const int lane = tid & 31;
  const int wave = tid >> 5;
  const int mt = wave & 3;            // wave M strip (0..3), 32 rows each
  const int nt = wave >> 2;           // wave N strip (0..1), 32 cols each
  v8f acc00 = {}, acc01 = {}, acc10 = {}, acc11 = {};
  for (int k0 = 0; k0 < K; k0 += 32) {
#if USE_TDM
    if (wave == 0) {                   // one DMA descriptor per tile
      tdm_load_tile32(&A[(size_t)bm * K + k0], K, &Asf[0][0], 128);
      tdm_load_tile32(&W[(size_t)bn * K + k0], K, &Bsf[0][0], 64);
      __builtin_amdgcn_s_wait_tensorcnt(0);
    }
#else
    // A tile: 128 rows x 32 floats = 1024 16B segments
    for (int i = tid; i < 1024; i += 256) {
      int r = i >> 3, s = (i & 7) << 2;
      copy16_g2l(&A[(size_t)(bm + r) * K + k0 + s], &Asf[r][s]);
    }
    // B tile: 64 rows x 32 floats = 512 16B segments
    for (int i = tid; i < 512; i += 256) {
      int r = i >> 3, s = (i & 7) << 2;
      copy16_g2l(&W[(size_t)(bn + r) * K + k0 + s], &Bsf[r][s]);
    }
    wait_async_lds();
#endif
    if (k0 + 32 < K) {                 // prefetch next K chunk
      if (tid < 128) __builtin_prefetch(&A[(size_t)(bm + tid) * K + k0 + 32], 0, 0);
      else if (tid < 192) __builtin_prefetch(&W[(size_t)(bn + tid - 128) * K + k0 + 32], 0, 0);
    }
    __syncthreads();
    V16BF a0, a1, b0, b1;
    const int lm = (mt << 5) + (lane & 15);
    const int ln = (nt << 5) + (lane & 15);
#pragma unroll
    for (int e = 0; e < 16; ++e) {
      int kk = wmma_kidx(lane, e);
      a0.u[e] = f2bf(Asf[lm][kk]);
      a1.u[e] = f2bf(Asf[lm + 16][kk]);
      b0.u[e] = f2bf(Bsf[ln][kk]);
      b1.u[e] = f2bf(Bsf[ln + 16][kk]);
    }
    acc00 = wmma_bf16(a0.v, b0.v, acc00);
    acc01 = wmma_bf16(a0.v, b1.v, acc01);
    acc10 = wmma_bf16(a1.v, b0.v, acc10);
    acc11 = wmma_bf16(a1.v, b1.v, acc11);
    __syncthreads();
  }
  const int n0 = bn + (nt << 5) + (lane & 15);
  const int m0 = bm + (mt << 5) + ((lane & 16) ? 8 : 0);
  const float bz0 = bias ? bias[n0] : 0.f;
  const float bz1 = bias ? bias[n0 + 16] : 0.f;
#pragma unroll
  for (int r = 0; r < 8; ++r) {
    float x00 = acc00[r] + bz0, x01 = acc01[r] + bz1;
    float x10 = acc10[r] + bz0, x11 = acc11[r] + bz1;
    if (act) {
      x00 = fmaxf(x00, 0.f); x01 = fmaxf(x01, 0.f);
      x10 = fmaxf(x10, 0.f); x11 = fmaxf(x11, 0.f);
    }
    C[(size_t)(m0 + r) * N + n0]           = x00;
    C[(size_t)(m0 + r) * N + n0 + 16]      = x01;
    C[(size_t)(m0 + 16 + r) * N + n0]      = x10;
    C[(size_t)(m0 + 16 + r) * N + n0 + 16] = x11;
  }
}

// ---------------------------------------------------------------------------
// Self-attention per (window n, head g). 128 threads = 4 waves.
// scores = (Q*scale)(64x32) @ K^T -> softmax -> @ V(64x32). All WMMA.
// ---------------------------------------------------------------------------
__global__ void __launch_bounds__(128)
self_attn(const float* __restrict__ qkproj, const float* __restrict__ vproj,
          float* __restrict__ obuf) {
  const int n = blockIdx.x;   // 0..255
  const int g = blockIdx.y;   // 0..7
  __shared__ unsigned short qs[64][32];
  __shared__ unsigned short ks[64][32];
  __shared__ float          sc[64][64];
  __shared__ unsigned short ps[64][64];
  __shared__ unsigned short vs[64][32];
  const int tid = threadIdx.x, lane = tid & 31, wave = tid >> 5;
  const float scale = 0.17677669529663689f;  // 1/sqrt(32)
  for (int i = tid; i < 2048; i += 128) {
    int l = i >> 5, d = i & 31;
    size_t base = (size_t)l * 256 + n;
    qs[l][d] = f2bf(qkproj[base * 512 + g * 32 + d] * scale);
    ks[l][d] = f2bf(qkproj[base * 512 + 256 + g * 32 + d]);
    vs[l][d] = f2bf(vproj[base * 256 + g * 32 + d]);
  }
  __syncthreads();
  {
    const int mr = (wave << 4) + (lane & 15);
    V16BF a;
#pragma unroll
    for (int e = 0; e < 16; ++e) a.u[e] = qs[mr][wmma_kidx(lane, e)];
    const int mb = (wave << 4) + ((lane & 16) ? 8 : 0);
    for (int ntile = 0; ntile < 4; ++ntile) {
      V16BF b;
#pragma unroll
      for (int e = 0; e < 16; ++e)
        b.u[e] = ks[(ntile << 4) + (lane & 15)][wmma_kidx(lane, e)];
      v8f acc = {};
      acc = wmma_bf16(a.v, b.v, acc);
      const int col = (ntile << 4) + (lane & 15);
#pragma unroll
      for (int r = 0; r < 8; ++r) sc[mb + r][col] = acc[r];
    }
  }
  __syncthreads();
  if (tid < 64) {
    float mx = -3.4e38f;
    for (int j = 0; j < 64; ++j) mx = fmaxf(mx, sc[tid][j]);
    float s = 0.f;
    for (int j = 0; j < 64; ++j) { float e = __expf(sc[tid][j] - mx); sc[tid][j] = e; s += e; }
    float inv = 1.f / s;
    for (int j = 0; j < 64; ++j) ps[tid][j] = f2bf(sc[tid][j] * inv);
  }
  __syncthreads();
  {
    v8f acc0 = {}, acc1 = {};
    const int mr = (wave << 4) + (lane & 15);
    for (int ks0 = 0; ks0 < 64; ks0 += 32) {
      V16BF a, b0, b1;
#pragma unroll
      for (int e = 0; e < 16; ++e) {
        int kk = wmma_kidx(lane, e);
        a.u[e]  = ps[mr][ks0 + kk];
        b0.u[e] = vs[ks0 + kk][lane & 15];
        b1.u[e] = vs[ks0 + kk][16 + (lane & 15)];
      }
      acc0 = wmma_bf16(a.v, b0.v, acc0);
      acc1 = wmma_bf16(a.v, b1.v, acc1);
    }
    const int mb = (wave << 4) + ((lane & 16) ? 8 : 0);
    const int d0 = lane & 15;
#pragma unroll
    for (int r = 0; r < 8; ++r) {
      size_t row = ((size_t)(mb + r) * 256 + n) * 256 + g * 32;
      obuf[row + d0]      = acc0[r];
      obuf[row + 16 + d0] = acc1[r];
    }
  }
}

// ---------------------------------------------------------------------------
// RCDA attention per (window n, head g). 128 threads = 4 waves.
// s_r = scale*Qr@Kr^T (64x16), s_c likewise; mask; softmax;
// w2[l, y*16+x] = a_c[l,y]*a_r[l,x]; out = w2(64x256) @ V(256x32) via WMMA.
// ---------------------------------------------------------------------------
__device__ __forceinline__ void softmax16(float* row) {
  float mx = row[0];
  for (int j = 1; j < 16; ++j) mx = fmaxf(mx, row[j]);
  float s = 0.f;
  for (int j = 0; j < 16; ++j) { float e = __expf(row[j] - mx); row[j] = e; s += e; }
  float inv = 1.f / s;
  for (int j = 0; j < 16; ++j) row[j] *= inv;
}

__global__ void __launch_bounds__(128)
rcda_attn(const float* __restrict__ qrp, const float* __restrict__ qcp,
          const float* __restrict__ krp, const float* __restrict__ kcp,
          const float* __restrict__ vb, const unsigned char* __restrict__ mrow,
          const unsigned char* __restrict__ mcol, float* __restrict__ outb) {
  const int n = blockIdx.x, g = blockIdx.y;
  __shared__ float          krS[16][32];
  __shared__ float          kcS[16][32];
  __shared__ float          arS[64][16];
  __shared__ float          acS[64][16];
  __shared__ unsigned short w2[64][256];
  __shared__ unsigned short vS[256][32];
  const int tid = threadIdx.x, lane = tid & 31, wave = tid >> 5;
  const float scale = 0.17677669529663689f;
  for (int i = tid; i < 512; i += 128) {
    int r = i >> 5, d = i & 31;
    krS[r][d] = krp[((size_t)n * 16 + r) * 256 + g * 32 + d];
    kcS[r][d] = kcp[((size_t)n * 16 + r) * 256 + g * 32 + d];
  }
  for (int i = tid; i < 8192; i += 128) {
    int p = i >> 5, d = i & 31;
    vS[p][d] = f2bf(vb[((size_t)n * 256 + p) * 256 + g * 32 + d]);
  }
  __syncthreads();
  {
    const int l = tid >> 1, h = tid & 1;
    float q[32];
    const size_t qbase = ((size_t)n * 64 + l) * 256 + g * 32;
#pragma unroll
    for (int d = 0; d < 32; ++d) q[d] = qrp[qbase + d];
    for (int x = h * 8; x < h * 8 + 8; ++x) {
      float s = 0.f;
#pragma unroll
      for (int d = 0; d < 32; ++d) s += q[d] * krS[x][d];
      arS[l][x] = mrow[n * 16 + x] ? -3.0e38f : s * scale;
    }
#pragma unroll
    for (int d = 0; d < 32; ++d) q[d] = qcp[qbase + d];
    for (int y = h * 8; y < h * 8 + 8; ++y) {
      float s = 0.f;
#pragma unroll
      for (int d = 0; d < 32; ++d) s += q[d] * kcS[y][d];
      acS[l][y] = mcol[n * 16 + y] ? -3.0e38f : s * scale;
    }
  }
  __syncthreads();
  if (tid < 64) { softmax16(arS[tid]); softmax16(acS[tid]); }
  __syncthreads();
  for (int i = tid; i < 16384; i += 128) {
    int l = i >> 8, p = i & 255;
    w2[l][p] = f2bf(acS[l][p >> 4] * arS[l][p & 15]);
  }
  __syncthreads();
  {
    v8f acc0 = {}, acc1 = {};
    const int mr = (wave << 4) + (lane & 15);
    for (int k0 = 0; k0 < 256; k0 += 32) {
      V16BF a, b0, b1;
#pragma unroll
      for (int e = 0; e < 16; ++e) {
        int kk = wmma_kidx(lane, e);
        a.u[e]  = w2[mr][k0 + kk];
        b0.u[e] = vS[k0 + kk][lane & 15];
        b1.u[e] = vS[k0 + kk][16 + (lane & 15)];
      }
      acc0 = wmma_bf16(a.v, b0.v, acc0);
      acc1 = wmma_bf16(a.v, b1.v, acc1);
    }
    const int mb = (wave << 4) + ((lane & 16) ? 8 : 0);
    const int d0 = lane & 15;
#pragma unroll
    for (int r = 0; r < 8; ++r) {
      size_t row = ((size_t)n * 64 + mb + r) * 256 + g * 32;
      outb[row + d0]      = acc0[r];
      outb[row + 16 + d0] = acc1[r];
    }
  }
}

// ---------------------------------------------------------------------------
// Elementwise / reduction helpers
// ---------------------------------------------------------------------------
__global__ void __launch_bounds__(256)
add2(const float* __restrict__ a, const float* __restrict__ b,
     float* __restrict__ c, int n) {
  int i = blockIdx.x * 256 + threadIdx.x;
  if (i < n) c[i] = a[i] + b[i];
}

// out(row) = LN(X(row) + Y(yrow)) * gamma + beta; rows are (l,n)-major.
__global__ void __launch_bounds__(256)
resid_ln(const float* __restrict__ X, const float* __restrict__ Y,
         const float* __restrict__ gamma, const float* __restrict__ beta,
         float* __restrict__ out, int transposedY) {
  const int rid = blockIdx.x;
  const int c = threadIdx.x;
  __shared__ float red[256];
  __shared__ float s_mean, s_rstd;
  const int l = rid >> 8, n = rid & 255;
  const size_t yrow = transposedY ? ((size_t)n * 64 + l) : (size_t)rid;
  float v = X[(size_t)rid * 256 + c] + Y[yrow * 256 + c];
  red[c] = v;
  __syncthreads();
  for (int s = 128; s > 0; s >>= 1) { if (c < s) red[c] += red[c + s]; __syncthreads(); }
  if (c == 0) s_mean = red[0] * (1.f / 256.f);
  __syncthreads();
  const float mean = s_mean;
  const float d = v - mean;
  red[c] = d * d;
  __syncthreads();
  for (int s = 128; s > 0; s >>= 1) { if (c < s) red[c] += red[c + s]; __syncthreads(); }
  if (c == 0) s_rstd = rsqrtf(red[0] * (1.f / 256.f) + 1e-5f);
  __syncthreads();
  out[(size_t)rid * 256 + c] = d * s_rstd * gamma[c] + beta[c];
}

// pos2posemb1d: out[q,2j]=sin(pos*2pi/t_j), out[q,2j+1]=cos(...), t_j=10000^(j/128)
__global__ void __launch_bounds__(256)
posemb1d(const float* __restrict__ ref, int axis, float* __restrict__ pe) {
  const int q = blockIdx.x;
  const int i = threadIdx.x;
  const float pos = ref[q * 2 + axis] * 6.283185307179586f;
  const float t = __powf(10000.f, (float)(i >> 1) * (1.f / 128.f));
  const float v = pos / t;
  pe[(size_t)q * 256 + i] = (i & 1) ? __cosf(v) : __sinf(v);
}

// qr_in[(n,l),c] = tgt1[(l,n),c] + ax[qi,c];  qc_in with ay.
__global__ void __launch_bounds__(256)
q_gather_add(const float* __restrict__ tgt1, const float* __restrict__ ax,
             const float* __restrict__ ay, const int* __restrict__ v_idx,
             float* __restrict__ qr_in, float* __restrict__ qc_in) {
  const int rid = blockIdx.x;            // n*64 + l
  const int n = rid >> 6, l = rid & 63;
  const int wn = v_idx[n];
  const int wr = (wn >> 3) & 7, wc = wn & 7;
  const int qi = (wr * 8 + (l >> 3)) * 64 + (wc * 8 + (l & 7));
  const int c = threadIdx.x;
  const float t = tgt1[((size_t)l * 256 + n) * 256 + c];
  qr_in[(size_t)rid * 256 + c] = t + ax[(size_t)qi * 256 + c];
  qc_in[(size_t)rid * 256 + c] = t + ay[(size_t)qi * 256 + c];
}

// window partition of srcs (B,C,H,W) -> (n, y, x, c)
__global__ void __launch_bounds__(256)
win_part(const float* __restrict__ srcs, const int* __restrict__ v_idx,
         float* __restrict__ wsrc) {
  const int bid = blockIdx.x;            // n*256 + ch
  const int n = bid >> 8, ch = bid & 255;
  const int wn = v_idx[n];
  const int b = wn >> 6, wr = (wn >> 3) & 7, wc = wn & 7;
  const int t = threadIdx.x, y = t >> 4, x = t & 15;
  const float v = srcs[(((size_t)b * 256 + ch) * 128 + (wr * 16 + y)) * 128 + (wc * 16 + x)];
  wsrc[(((size_t)n * 16 + y) * 16 + x) * 256 + ch] = v;
}

// kr_in[(n,x),c] = mean_y wsrc[n,y,x,c] + posemb_row[b, wc*16+x, c]
__global__ void __launch_bounds__(256)
krow_mean(const float* __restrict__ wsrc, const float* __restrict__ prow,
          const int* __restrict__ v_idx, float* __restrict__ krin) {
  const int bid = blockIdx.x;            // n*16 + x
  const int n = bid >> 4, x = bid & 15;
  const int wn = v_idx[n];
  const int b = wn >> 6, wc = wn & 7;
  const int c = threadIdx.x;
  float s = 0.f;
  for (int y = 0; y < 16; ++y)
    s += wsrc[(((size_t)n * 16 + y) * 16 + x) * 256 + c];
  s *= (1.f / 16.f);
  s += prow[((size_t)b * 128 + wc * 16 + x) * 256 + c];
  krin[(size_t)bid * 256 + c] = s;
}

// kc_in[(n,y),c] = mean_x wsrc[n,y,x,c] + posemb_col[b, wr*16+y, c]
__global__ void __launch_bounds__(256)
kcol_mean(const float* __restrict__ wsrc, const float* __restrict__ pcol,
          const int* __restrict__ v_idx, float* __restrict__ kcin) {
  const int bid = blockIdx.x;            // n*16 + y
  const int n = bid >> 4, y = bid & 15;
  const int wn = v_idx[n];
  const int b = wn >> 6, wr = (wn >> 3) & 7;
  const int c = threadIdx.x;
  float s = 0.f;
  for (int x = 0; x < 16; ++x)
    s += wsrc[(((size_t)n * 16 + y) * 16 + x) * 256 + c];
  s *= (1.f / 16.f);
  s += pcol[((size_t)b * 128 + wr * 16 + y) * 256 + c];
  kcin[(size_t)bid * 256 + c] = s;
}

__global__ void __launch_bounds__(256)
make_masks(const unsigned char* __restrict__ pad, const int* __restrict__ v_idx,
           unsigned char* __restrict__ mrow, unsigned char* __restrict__ mcol) {
  const int i = blockIdx.x * 256 + threadIdx.x;   // 0..8191
  if (i >= 8192) return;
  const int half = i >> 12;
  const int j = i & 4095;                         // n*16 + t
  const int n = j >> 4, t = j & 15;
  const int wn = v_idx[n];
  const int b = wn >> 6, wr = (wn >> 3) & 7, wc = wn & 7;
  if (half == 0)
    mrow[j] = pad[((size_t)b * 128 + wr * 16) * 128 + (wc * 16 + t)];
  else
    mcol[j] = pad[((size_t)b * 128 + wr * 16 + t) * 128 + (wc * 16)];
}

// ---------------------------------------------------------------------------
// Host launcher
// ---------------------------------------------------------------------------
extern "C" void kernel_launch(void* const* d_in, const int* in_sizes, int n_in,
                              void* d_out, int out_size, void* d_ws, size_t ws_size,
                              hipStream_t stream) {
  (void)in_sizes; (void)n_in; (void)out_size; (void)ws_size;
  // setup_inputs() dict order
  const float* tgt        = (const float*)d_in[0];
  const float* query_pos  = (const float*)d_in[1];
  const float* refpts     = (const float*)d_in[2];
  const float* srcs       = (const float*)d_in[3];
  /* d_in[4] srcs_pos: unused by the reference forward */
  const float* posemb_row = (const float*)d_in[5];
  const float* posemb_col = (const float*)d_in[6];
  const unsigned char* padmask = (const unsigned char*)d_in[7];
  const int*   v_idx      = (const int*)d_in[8];
  /* 9..12: win_h, win_w, q_h, q_w scalars (hardcoded 16,16,64,64) */
  const float* ap1_w1 = (const float*)d_in[13];
  const float* ap1_b1 = (const float*)d_in[14];
  const float* ap1_w2 = (const float*)d_in[15];
  const float* ap1_b2 = (const float*)d_in[16];
  const float* sa_in_w  = (const float*)d_in[17];
  const float* sa_in_b  = (const float*)d_in[18];
  const float* sa_out_w = (const float*)d_in[19];
  const float* sa_out_b = (const float*)d_in[20];
  const float* ca_qr_w = (const float*)d_in[21];
  const float* ca_qr_b = (const float*)d_in[22];
  const float* ca_qc_w = (const float*)d_in[23];
  const float* ca_qc_b = (const float*)d_in[24];
  const float* ca_kr_w = (const float*)d_in[25];
  const float* ca_kr_b = (const float*)d_in[26];
  const float* ca_kc_w = (const float*)d_in[27];
  const float* ca_kc_b = (const float*)d_in[28];
  const float* ca_v_w  = (const float*)d_in[29];
  const float* ca_v_b  = (const float*)d_in[30];
  const float* ca_out_w = (const float*)d_in[31];
  const float* ca_out_b = (const float*)d_in[32];
  const float* n1_g = (const float*)d_in[33];
  const float* n1_b = (const float*)d_in[34];
  const float* n2_g = (const float*)d_in[35];
  const float* n2_b = (const float*)d_in[36];
  const float* ffn_w1 = (const float*)d_in[37];
  const float* ffn_b1 = (const float*)d_in[38];
  const float* ffn_w2 = (const float*)d_in[39];
  const float* ffn_b2 = (const float*)d_in[40];
  const float* ffn_ng = (const float*)d_in[41];
  const float* ffn_nb = (const float*)d_in[42];
  float* out = (float*)d_out;

  // Bump allocator over d_ws (256B aligned)
  size_t off = 0;
  auto alloc = [&](size_t bytes) -> char* {
    off = (off + 255) & ~(size_t)255;
    char* p = (char*)d_ws + off;
    off += bytes;
    return p;
  };
  const size_t F = sizeof(float);
  const size_t M1 = 16384;    // LQ*NSEL
  float* qkbuf  = (float*)alloc(M1 * 256 * F);
  float* qkproj = (float*)alloc(M1 * 512 * F);
  float* vproj  = (float*)alloc(M1 * 256 * F);
  float* tgt1   = (float*)alloc(M1 * 256 * F);
  float* ax     = (float*)alloc(4096 * 256 * F);
  float* ay     = (float*)alloc(4096 * 256 * F);
  float* qr_in  = (float*)alloc(M1 * 256 * F);
  float* qc_in  = (float*)alloc(M1 * 256 * F);
  float* wsrc   = (float*)alloc((size_t)65536 * 256 * F);
  float* krin   = (float*)alloc(4096 * 256 * F);
  float* kcin   = (float*)alloc(4096 * 256 * F);
  float* vbig   = (float*)alloc((size_t)65536 * 256 * F);
  float* qrproj = (float*)alloc(M1 * 256 * F);
  float* qcproj = (float*)alloc(M1 * 256 * F);
  float* krproj = (float*)alloc(4096 * 256 * F);
  float* kcproj = (float*)alloc(4096 * 256 * F);
  float* rcdao  = (float*)alloc(M1 * 256 * F);
  float* tgt2   = (float*)alloc(M1 * 256 * F);
  unsigned char* mrow = (unsigned char*)alloc(4096);
  unsigned char* mcol = (unsigned char*)alloc(4096);
  // Aliases (lifetimes disjoint):
  float* obuf    = qkbuf;                    // after qkv GEMMs consume qkbuf
  float* attnout = qkproj;                   // after self_attn consumes qkproj
  float* pebuf   = qkproj + M1 * 256;        // inside qkproj region
  float* hidbuf  = pebuf + 4096 * 256;
  float* caout   = qr_in;                    // after qr projection
  float* ffnout  = qc_in;                    // after qc projection
  float* ffnh    = wsrc;                     // 16384*1024 == 65536*256 floats

  const dim3 B256(256);
  auto gemm = [&](const float* A, const float* W, const float* bias, float* C,
                  int M, int N, int K, int act) {
    dim3 grid(N / 64, M / 128);
    hipLaunchKernelGGL(gemm_bias_act, grid, B256, 0, stream, A, W, bias, C, M, N, K, act);
  };

  // ---- self attention branch ----
  hipLaunchKernelGGL(add2, dim3(16384), B256, 0, stream, tgt, query_pos, qkbuf,
                     (int)(M1 * 256));
  gemm(qkbuf, sa_in_w, sa_in_b, qkproj, (int)M1, 512, 256, 0);          // Q|K proj
  gemm(tgt, sa_in_w + (size_t)512 * 256, sa_in_b + 512, vproj, (int)M1, 256, 256, 0);
  hipLaunchKernelGGL(self_attn, dim3(256, 8), dim3(128), 0, stream, qkproj, vproj, obuf);
  gemm(obuf, sa_out_w, sa_out_b, attnout, (int)M1, 256, 256, 0);
  hipLaunchKernelGGL(resid_ln, dim3(16384), B256, 0, stream, tgt, attnout, n2_g, n2_b,
                     tgt1, 0);

  // ---- query positional embeddings (adapt1d o pos2posemb1d) ----
  hipLaunchKernelGGL(posemb1d, dim3(4096), B256, 0, stream, refpts, 0, pebuf);
  gemm(pebuf, ap1_w1, ap1_b1, hidbuf, 4096, 256, 256, 1);
  gemm(hidbuf, ap1_w2, ap1_b2, ax, 4096, 256, 256, 0);
  hipLaunchKernelGGL(posemb1d, dim3(4096), B256, 0, stream, refpts, 1, pebuf);
  gemm(pebuf, ap1_w1, ap1_b1, hidbuf, 4096, 256, 256, 1);
  gemm(hidbuf, ap1_w2, ap1_b2, ay, 4096, 256, 256, 0);
  hipLaunchKernelGGL(q_gather_add, dim3(16384), B256, 0, stream, tgt1, ax, ay, v_idx,
                     qr_in, qc_in);

  // ---- source window partition, key means, masks ----
  hipLaunchKernelGGL(win_part, dim3(65536), B256, 0, stream, srcs, v_idx, wsrc);
  hipLaunchKernelGGL(krow_mean, dim3(4096), B256, 0, stream, wsrc, posemb_row, v_idx, krin);
  hipLaunchKernelGGL(kcol_mean, dim3(4096), B256, 0, stream, wsrc, posemb_col, v_idx, kcin);
  hipLaunchKernelGGL(make_masks, dim3(32), B256, 0, stream, padmask, v_idx, mrow, mcol);

  // ---- RCDA projections + attention ----
  gemm(qr_in, ca_qr_w, ca_qr_b, qrproj, (int)M1, 256, 256, 0);
  gemm(qc_in, ca_qc_w, ca_qc_b, qcproj, (int)M1, 256, 256, 0);
  gemm(krin,  ca_kr_w, ca_kr_b, krproj, 4096, 256, 256, 0);
  gemm(kcin,  ca_kc_w, ca_kc_b, kcproj, 4096, 256, 256, 0);
  gemm(wsrc,  ca_v_w,  ca_v_b,  vbig,  65536, 256, 256, 0);
  hipLaunchKernelGGL(rcda_attn, dim3(256, 8), dim3(128), 0, stream,
                     qrproj, qcproj, krproj, kcproj, vbig, mrow, mcol, rcdao);
  gemm(rcdao, ca_out_w, ca_out_b, caout, (int)M1, 256, 256, 0);
  hipLaunchKernelGGL(resid_ln, dim3(16384), B256, 0, stream, tgt1, caout, n1_g, n1_b,
                     tgt2, 1 /* caout rows are (n,l)-major */);

  // ---- FFN + final LN ----
  gemm(tgt2, ffn_w1, ffn_b1, ffnh, (int)M1, 1024, 256, 1);
  gemm(ffnh, ffn_w2, ffn_b2, ffnout, (int)M1, 256, 1024, 0);
  hipLaunchKernelGGL(resid_ln, dim3(16384), B256, 0, stream, tgt2, ffnout, ffn_ng, ffn_nb,
                     out, 0);
}
